// EdgeNet_61856118997067
// MI455X (gfx1250) — compile-verified, matrix operand order checked by reading, not run on hardware
//
#include <hip/hip_runtime.h>
#include <hip/hip_bf16.h>

// ---------- vector typedefs (POD ext_vector types, safe in unions) ----------
typedef __attribute__((ext_vector_type(16))) __bf16       v16bf;
typedef __attribute__((ext_vector_type(8)))  float        v8f;
typedef __attribute__((ext_vector_type(4)))  unsigned int uint4v;
typedef __attribute__((ext_vector_type(4)))  float        float4v;

__device__ __forceinline__ unsigned short f2bf(float f) {
  unsigned u = __float_as_uint(f);
  return (unsigned short)((u + 0x7FFFu + ((u >> 16) & 1u)) >> 16);  // RNE
}

union FragBF { uint4v u[2]; v16bf v; };

// ======================= WMMA GEMM: C[M x 128] = A[M x K] * B[K x 128] ======
// A: bf16 row-major [M][K].  Bt: bf16 row-major [128][K] (B transposed).
// Compile-time epilogue: + bias[col], * rowscale[row] (SCALE), relu (RELU),
// optional bf16 copy (CBF). Branch-free per instantiation.
// Block = 256 threads = 8 waves; wave w owns 16x16 tile (blockIdx.x, w).
template<int K, bool SCALE, bool CBF, bool RELU>
__global__ void __launch_bounds__(256) k_gemm_bf16(
    const unsigned short* __restrict__ A,
    const unsigned short* __restrict__ Bt,
    const float* __restrict__ bias,
    const float* __restrict__ rowscale,
    float* __restrict__ C,
    unsigned short* __restrict__ Cbf)
{
  const int lane = threadIdx.x & 31;
  const int wave = threadIdx.x >> 5;
  const int lcol = lane & 15;
  const int lhi  = lane >> 4;                    // 0: K 0-7,16-23; 1: K 8-15,24-31 (per 32-chunk)
  const long rowbase = (long)blockIdx.x * 16;
  const int  colbase = wave * 16;

  // A lane row (M = lcol for both half-waves), Bt lane row (N = lcol)
  const unsigned short* arow = A  + (size_t)(rowbase + lcol) * K;
  const unsigned short* brow = Bt + (size_t)(colbase + lcol) * K;

  v8f acc = {0.f, 0.f, 0.f, 0.f, 0.f, 0.f, 0.f, 0.f};

#pragma unroll
  for (int kk = 0; kk < K; kk += 32) {
    const int kb = kk + lhi * 8;                 // elems 0..7  -> K=kb..kb+7
    FragBF fa, fb;                               // elems 8..15 -> K=kb+16..kb+23
    fa.u[0] = *(const uint4v*)(arow + kb);
    fa.u[1] = *(const uint4v*)(arow + kb + 16);
    fb.u[0] = *(const uint4v*)(brow + kb);
    fb.u[1] = *(const uint4v*)(brow + kb + 16);
    acc = __builtin_amdgcn_wmma_f32_16x16x32_bf16(
        false, fa.v, false, fb.v, (short)0, acc, false, false);
  }

  const int ccol = colbase + lcol;
  const float bc = bias[ccol];
  float vals[8];
#pragma unroll
  for (int r = 0; r < 8; ++r) {
    const long row = rowbase + lhi * 8 + r;      // C layout: lanes 16-31 -> M=8+r
    float val = acc[r] + bc;
    if (SCALE) val *= rowscale[row];
    if (RELU)  val = fmaxf(val, 0.0f);
    vals[r] = val;
  }
#pragma unroll
  for (int r = 0; r < 8; ++r) {
    const long row = rowbase + lhi * 8 + r;
    C[row * 128 + ccol] = vals[r];
  }
  if (CBF) {
#pragma unroll
    for (int r = 0; r < 8; ++r) {
      const long row = rowbase + lhi * 8 + r;
      Cbf[row * 128 + ccol] = f2bf(vals[r]);
    }
  }
}

// ======================= small utility kernels ==============================
__global__ void k_fill1(float* p, long n) {
  long t = (long)blockIdx.x * blockDim.x + threadIdx.x;
  if (t < n) p[t] = 1.0f;
}
__global__ void k_rsqrt(float* p, long n) {
  long t = (long)blockIdx.x * blockDim.x + threadIdx.x;
  if (t < n) p[t] = rsqrtf(fmaxf(p[t], 1.0f));
}
__global__ void k_count(const int* __restrict__ snd, const int* __restrict__ rcv,
                        float* ds, float* dr, int nE) {
  long t = (long)blockIdx.x * blockDim.x + threadIdx.x;
  if (t >= nE) return;
  atomicAdd(ds + snd[t], 1.0f);
  atomicAdd(dr + rcv[t], 1.0f);
}
// weight transpose + bf16:  W[Kin][Cout] (fp32) -> Wt[Cout][Kin] (bf16)
__global__ void k_wt(const float* __restrict__ W, unsigned short* __restrict__ Wt,
                     int Kin, int Cout) {
  long t = (long)blockIdx.x * blockDim.x + threadIdx.x;
  if (t >= (long)Kin * Cout) return;
  int k = (int)(t / Cout), c = (int)(t % Cout);
  Wt[(size_t)c * Kin + k] = f2bf(W[(size_t)k * Cout + c]);
}
__global__ void k_embed(const int* __restrict__ nodes, const float* __restrict__ embed,
                        float* __restrict__ h, unsigned short* __restrict__ hbf, long total) {
  long t = (long)blockIdx.x * blockDim.x + threadIdx.x;
  if (t >= total) return;
  long n = t >> 7; int c = (int)(t & 127);
  float x = embed[(size_t)nodes[n] * 128 + c];
  h[t] = x; hbf[t] = f2bf(x);
}
__global__ void k_copy4(float4v* __restrict__ dst, const float4v* __restrict__ src, long n4) {
  long t = (long)blockIdx.x * blockDim.x + threadIdx.x;
  if (t < n4) dst[t] = src[t];
}
// 32 lanes per edge, 4 channels/lane: agg[rcv] += msg[snd]
__global__ void k_scatter(const float* __restrict__ msg, const int* __restrict__ snd,
                          const int* __restrict__ rcv, float* __restrict__ agg, int nE) {
  long t = (long)blockIdx.x * blockDim.x + threadIdx.x;
  int e = (int)(t >> 5);
  if (e >= nE) return;
  int lane = (int)(t & 31);
  int s = snd[e], r = rcv[e];
  float4v m = *(const float4v*)(msg + (size_t)s * 128 + lane * 4);
  float* dst = agg + (size_t)r * 128 + lane * 4;
  atomicAdd(dst + 0, m.x); atomicAdd(dst + 1, m.y);
  atomicAdd(dst + 2, m.z); atomicAdd(dst + 3, m.w);
}
// cat[n][0:128) = bf16(aggA*r1[n]); cat[n][128:256) = bf16(aggG*r2[n])
__global__ void k_concat(const float* __restrict__ aggA, const float* __restrict__ aggG,
                         const float* __restrict__ r1, const float* __restrict__ r2,
                         unsigned short* __restrict__ cat, long total) {
  long t = (long)blockIdx.x * blockDim.x + threadIdx.x;
  if (t >= total) return;
  long n = t >> 7; int c = (int)(t & 127);
  cat[n * 256 + c]       = f2bf(aggA[t] * r1[n]);
  cat[n * 256 + 128 + c] = f2bf(aggG[t] * r2[n]);
}
// logits[e] = dot(lg[snd[e]], lg[rcv[e]]), one wave per edge
__global__ void k_edgedot(const float* __restrict__ lg, const int* __restrict__ snd,
                          const int* __restrict__ rcv, float* __restrict__ out, int nE) {
  long t = (long)blockIdx.x * blockDim.x + threadIdx.x;
  int e = (int)(t >> 5);
  if (e >= nE) return;
  int lane = (int)(t & 31);
  int s = snd[e], r = rcv[e];
  float4v a = *(const float4v*)(lg + (size_t)s * 128 + lane * 4);
  float4v b = *(const float4v*)(lg + (size_t)r * 128 + lane * 4);
  float d = a.x * b.x + a.y * b.y + a.z * b.z + a.w * b.w;
  for (int o = 16; o > 0; o >>= 1) d += __shfl_down(d, o, 32);
  if (lane == 0) out[e] = d;
}
// graph pooling: mean over nodes 1..npg-1 of h
__global__ void k_pool(const float* __restrict__ h, float* __restrict__ v,
                       unsigned short* __restrict__ vbf, int npg) {
  int b = blockIdx.x, c = threadIdx.x;                 // 128 threads/graph
  const float* base = h + (size_t)b * npg * 128;
  float s = 0.f;
  for (int i = 1; i < npg; ++i) s += base[(size_t)i * 128 + c];
  s /= (float)(npg - 1);
  v[(size_t)b * 128 + c] = s;
  vbf[(size_t)b * 128 + c] = f2bf(s);
}
// out[b] = tanh(dot(v[b], w_out) + b_out), one wave per graph
__global__ void k_head(const float* __restrict__ v, const float* __restrict__ w,
                       const float* __restrict__ bo, float* __restrict__ out) {
  int b = blockIdx.x, lane = threadIdx.x;              // 32 threads
  float4v a = *(const float4v*)(v + (size_t)b * 128 + lane * 4);
  float4v c = *(const float4v*)(w + lane * 4);
  float s = a.x * c.x + a.y * c.y + a.z * c.z + a.w * c.w;
  for (int o = 16; o > 0; o >>= 1) s += __shfl_down(s, o, 32);
  if (lane == 0) out[b] = tanhf(s + bo[0]);
}

// ======================= host-side launch ===================================
static inline unsigned gb(long n, int t) { return (unsigned)((n + t - 1) / t); }
static inline char* carve(char*& p, size_t bytes) {
  char* r = p; p += (bytes + 255) & ~(size_t)255; return r;
}

extern "C" void kernel_launch(void* const* d_in, const int* in_sizes, int n_in,
                              void* d_out, int out_size, void* d_ws, size_t ws_size,
                              hipStream_t stream) {
  const int* nodes   = (const int*)d_in[0];
  const int* snd     = (const int*)d_in[1];
  const int* rcv     = (const int*)d_in[2];
  const int* gsnd    = (const int*)d_in[3];
  const int* grcv    = (const int*)d_in[4];
  const float* embed = (const float*)d_in[6];
  const float* w1 = (const float*)d_in[7],  *b1 = (const float*)d_in[8];
  const float* w2 = (const float*)d_in[9],  *b2 = (const float*)d_in[10];
  const float* w3 = (const float*)d_in[11], *b3 = (const float*)d_in[12];
  const float* wl = (const float*)d_in[13], *bl = (const float*)d_in[14];
  const float* ew = (const float*)d_in[15], *eb = (const float*)d_in[16];
  const float* wo = (const float*)d_in[17], *bo = (const float*)d_in[18];

  const long N  = in_sizes[0];
  const int  E  = in_sizes[1];
  const int  EG = in_sizes[3];
  const int  B  = in_sizes[5];
  const int  NPG = (int)(N / B);
  const long NF = N * 128;
  const int  NGNN = 7, NEVAL = 5;

  // ---- carve workspace ----
  char* p = (char*)d_ws;
  float*          h     = (float*)carve(p, NF * 4);
  unsigned short* hbf   = (unsigned short*)carve(p, NF * 2);
  float*          msg   = (float*)carve(p, NF * 4);          // also reused as lg
  float*          aggA  = (float*)carve(p, NF * 4);
  float*          aggG  = (float*)carve(p, NF * 4);
  unsigned short* catbf = (unsigned short*)carve(p, N * 256 * 2);
  float*          s1    = (float*)carve(p, N * 4);
  float*          r1    = (float*)carve(p, N * 4);
  float*          s2    = (float*)carve(p, N * 4);
  float*          r2    = (float*)carve(p, N * 4);
  unsigned short* w1t   = (unsigned short*)carve(p, (size_t)NGNN * 16384 * 2);
  unsigned short* w2t   = (unsigned short*)carve(p, (size_t)NGNN * 16384 * 2);
  unsigned short* w3t   = (unsigned short*)carve(p, (size_t)NGNN * 32768 * 2);
  unsigned short* wlt   = (unsigned short*)carve(p, 16384 * 2);
  unsigned short* ewt   = (unsigned short*)carve(p, (size_t)NEVAL * 16384 * 2);
  float*          vf0   = (float*)carve(p, (size_t)B * 128 * 4);
  float*          vf1   = (float*)carve(p, (size_t)B * 128 * 4);
  unsigned short* vb0   = (unsigned short*)carve(p, (size_t)B * 128 * 2);
  unsigned short* vb1   = (unsigned short*)carve(p, (size_t)B * 128 * 2);
  float* vf[2] = {vf0, vf1}; unsigned short* vb[2] = {vb0, vb1};
  (void)ws_size; (void)n_in; (void)out_size;

  // ---- weights -> bf16, transposed (once per launch) ----
  for (int i = 0; i < NGNN; ++i) {
    k_wt<<<gb(16384, 256), 256, 0, stream>>>(w1 + (size_t)i * 16384, w1t + (size_t)i * 16384, 128, 128);
    k_wt<<<gb(16384, 256), 256, 0, stream>>>(w2 + (size_t)i * 16384, w2t + (size_t)i * 16384, 128, 128);
    k_wt<<<gb(32768, 256), 256, 0, stream>>>(w3 + (size_t)i * 32768, w3t + (size_t)i * 32768, 256, 128);
  }
  k_wt<<<gb(16384, 256), 256, 0, stream>>>(wl, wlt, 128, 128);
  for (int i = 0; i < NEVAL; ++i)
    k_wt<<<gb(16384, 256), 256, 0, stream>>>(ew + (size_t)i * 16384, ewt + (size_t)i * 16384, 128, 128);

  // ---- degree norms (self-edge -> init 1); s1,r1,s2,r2 are contiguous ----
  k_fill1<<<gb(4 * N, 256), 256, 0, stream>>>(s1, 4 * N);
  k_count<<<gb(E, 256), 256, 0, stream>>>(snd, rcv, s1, r1, E);
  k_count<<<gb(EG, 256), 256, 0, stream>>>(gsnd, grcv, s2, r2, EG);
  k_rsqrt<<<gb(4 * N, 256), 256, 0, stream>>>(s1, 4 * N);

  // ---- embedding ----
  k_embed<<<gb(NF, 256), 256, 0, stream>>>(nodes, embed, h, hbf, NF);

  const unsigned gemmN = (unsigned)(N / 16);
  // ---- GNN layers ----
  for (int i = 0; i < NGNN; ++i) {
    // move-edge branch: msg = (h @ W1 + b1) * s1 ; aggA = msg(self) + scatter
    k_gemm_bf16<128, true, false, false><<<gemmN, 256, 0, stream>>>(
        hbf, w1t + (size_t)i * 16384, b1 + (size_t)i * 128, s1, msg, nullptr);
    k_copy4<<<gb(NF / 4, 256), 256, 0, stream>>>((float4v*)aggA, (const float4v*)msg, NF / 4);
    k_scatter<<<gb((long)E * 32, 256), 256, 0, stream>>>(msg, snd, rcv, aggA, E);
    // grid-edge branch
    k_gemm_bf16<128, true, false, false><<<gemmN, 256, 0, stream>>>(
        hbf, w2t + (size_t)i * 16384, b2 + (size_t)i * 128, s2, msg, nullptr);
    k_copy4<<<gb(NF / 4, 256), 256, 0, stream>>>((float4v*)aggG, (const float4v*)msg, NF / 4);
    k_scatter<<<gb((long)EG * 32, 256), 256, 0, stream>>>(msg, gsnd, grcv, aggG, EG);
    // concat (* r_norms) -> bf16, then h = relu(cat @ W3 + b3)
    k_concat<<<gb(NF, 256), 256, 0, stream>>>(aggA, aggG, r1, r2, catbf, NF);
    k_gemm_bf16<256, false, true, true><<<gemmN, 256, 0, stream>>>(
        catbf, w3t + (size_t)i * 32768, b3 + (size_t)i * 128, nullptr, h, hbf);
  }

  // ---- logit head: lg = h @ w_logit + b_logit ; logits[e] = <lg_s, lg_r> ----
  k_gemm_bf16<128, false, false, false><<<gemmN, 256, 0, stream>>>(
      hbf, wlt, bl, nullptr, msg, nullptr);
  k_edgedot<<<gb((long)E * 32, 256), 256, 0, stream>>>(msg, snd, rcv, (float*)d_out, E);

  // ---- eval head ----
  k_pool<<<(unsigned)B, 128, 0, stream>>>(h, vf[0], vb[0], NPG);
  for (int i = 0; i < NEVAL; ++i) {
    k_gemm_bf16<128, false, true, true><<<(unsigned)(B / 16), 256, 0, stream>>>(
        vb[i & 1], ewt + (size_t)i * 16384, eb + (size_t)i * 128, nullptr,
        vf[(i + 1) & 1], vb[(i + 1) & 1]);
  }
  k_head<<<(unsigned)B, 32, 0, stream>>>(vf[NEVAL & 1], wo, bo, (float*)d_out + E);
}
